// CausalSelfAttention_2594160246824
// MI455X (gfx1250) — matrix operand order repaired and checked by reading, not compiled, not run on hardware
//
#include <hip/hip_runtime.h>

// ---------------------------------------------------------------------------
// Causal self-attention forward for MI455X (gfx1250, wave32, WMMA).
//   B=4, S=2048, D=2048, H=16, hd=128, MAX_SEQ=2048, positions = arange(S).
// Pipeline: cvt(f32->bf16) -> QKV GEMM (WMMA bf16, async-LDS double buffer)
// -> RoPE/split (K stored transposed) -> flash attention (WMMA bf16, async-LDS
// K/V staging, online softmax, LDS P-transpose) -> output GEMM (f32 out).
// ---------------------------------------------------------------------------

#define B_   4
#define S_   2048
#define D_   2048
#define H_   16
#define HD_  128
#define BH_  (B_ * H_)
#define M_   (B_ * S_)     // 8192 rows for the big GEMMs
#define N3_  (3 * D_)      // 6144

typedef __attribute__((ext_vector_type(16))) __bf16 v16bf;
typedef __attribute__((ext_vector_type(8)))  float  v8f;
typedef __attribute__((ext_vector_type(4)))  int    v4i;

union FragB16 {            // 16 bf16 elements = 8 VGPRs (one WMMA A or B operand)
    uint4  u[2];
    v16bf  v;
};

__device__ __forceinline__ v8f zero8() {
    v8f z;
    #pragma unroll
    for (int i = 0; i < 8; ++i) z[i] = 0.0f;
    return z;
}

// ---------------------------------------------------------------------------
// Async global->LDS staging (gfx1250 GLOBAL_LOAD_ASYNC_TO_LDS_B128, ASYNCcnt).
// Builtin signature (from compiler diagnostic): arg0 is int4* in AS1 (global),
// non-const. LDS side passed as int4* in AS3. Integer casts build the
// exactly-typed pointers (low 32 bits of a generic LDS pointer are the LDS
// byte offset per the ISA aperture rule) and launder constness.
// ---------------------------------------------------------------------------
#if __has_builtin(__builtin_amdgcn_global_load_async_to_lds_b128) && \
    __has_builtin(__builtin_amdgcn_s_wait_asynccnt)
#define ASYNC_LDS 1
#else
#define ASYNC_LDS 0
#endif

__device__ __forceinline__ void cp16_g2l(const __bf16* g, __bf16* l) {
#if ASYNC_LDS
    typedef __attribute__((address_space(1))) v4i* gp_t;
    typedef __attribute__((address_space(3))) v4i* lp_t;
    __builtin_amdgcn_global_load_async_to_lds_b128(
        (gp_t)(uintptr_t)(const void*)g,
        (lp_t)(uint32_t)(uintptr_t)(void*)l, 0, 0);
#else
    *(uint4*)l = *(const uint4*)g;
#endif
}

__device__ __forceinline__ void stage_commit() {
#if ASYNC_LDS
    __builtin_amdgcn_s_wait_asynccnt(0);
#endif
    __syncthreads();
}

// ---------------------------------------------------------------------------
// Kernel 1: fp32 -> bf16 conversion (grid-stride)
// ---------------------------------------------------------------------------
__global__ void cvt_f32_to_bf16(const float* __restrict__ in,
                                __bf16* __restrict__ out, size_t n) {
    size_t i = (size_t)blockIdx.x * blockDim.x + threadIdx.x;
    size_t stride = (size_t)gridDim.x * blockDim.x;
    for (; i < n; i += stride) out[i] = (__bf16)in[i];
}

// ---------------------------------------------------------------------------
// Kernel 2/5: C[M,N] = A[M,K] @ B[K,N], bf16 in, f32 accum via
// v_wmma_f32_16x16x32_bf16. Block tile 128x128 (8 waves, wave tile 64x32),
// K-step 32, async-LDS double-buffered staging of A (128x32) and B (32x128).
// A fragment (16x32): lane m = lane&15, half = lane>>4;
//   K = half*8 + [0..7] and 16 + half*8 + [0..7].
// B fragment (32x16): lane holds column N = lane&15, 16 K in A's order.
// C/D (16x16 f32): VGPR v -> row v + 8*half, col = lane&15.
// ---------------------------------------------------------------------------
__global__ __launch_bounds__(256)
void gemm_bf16_wmma(const __bf16* __restrict__ A, const __bf16* __restrict__ Bm,
                    void* __restrict__ C, int M, int N, int K, int c_is_f32) {
    __shared__ __align__(16) __bf16 sA[2][128 * 32];   // 2 x 8KB
    __shared__ __align__(16) __bf16 sB[2][32 * 128];   // 2 x 8KB

    const int tid  = threadIdx.x;
    const int lane = tid & 31;
    const int wave = tid >> 5;
    const int wr   = wave >> 2;        // 0..1
    const int wc   = wave & 3;         // 0..3
    const int half = lane >> 4;
    const int ml   = lane & 15;

    const int blockM = blockIdx.y * 128;
    const int blockN = blockIdx.x * 128;
    const int waveM  = blockM + wr * 64;
    const int waveN  = blockN + wc * 32;

    v8f acc[4][2];
    #pragma unroll
    for (int i = 0; i < 4; ++i)
        #pragma unroll
        for (int j = 0; j < 2; ++j) acc[i][j] = zero8();

    // Stage one 128x32 A tile + 32x128 B tile (each 512 x 16B chunks; 2 per thread)
    auto stage = [&](int buf, int k0) {
        #pragma unroll
        for (int i = 0; i < 2; ++i) {
            const int chunk = tid * 2 + i;            // 0..511
            const int arow = chunk >> 2, ac = chunk & 3;
            cp16_g2l(A + (size_t)(blockM + arow) * K + k0 + ac * 8,
                     &sA[buf][arow * 32 + ac * 8]);
            const int brow = chunk >> 4, bc = chunk & 15;
            cp16_g2l(Bm + (size_t)(k0 + brow) * N + blockN + bc * 8,
                     &sB[buf][brow * 128 + bc * 8]);
        }
    };

    stage(0, 0);
    stage_commit();

    int buf = 0;
    for (int k0 = 0; k0 < K; k0 += 32) {
        if (k0 + 32 < K) stage(buf ^ 1, k0 + 32);   // overlaps the WMMAs below

        FragB16 a[4];
        #pragma unroll
        for (int ms = 0; ms < 4; ++ms) {
            const __bf16* ap = &sA[buf][(wr * 64 + ms * 16 + ml) * 32 + half * 8];
            a[ms].u[0] = *(const uint4*)(ap);
            a[ms].u[1] = *(const uint4*)(ap + 16);
        }
        FragB16 b[2];
        #pragma unroll
        for (int ns = 0; ns < 2; ++ns) {
            const __bf16* bp = &sB[buf][wc * 32 + ns * 16 + ml];
            #pragma unroll
            for (int s = 0; s < 16; ++s)
                b[ns].v[s] = bp[((s & 7) + half * 8 + (s >> 3) * 16) * 128];
        }
        #pragma unroll
        for (int ms = 0; ms < 4; ++ms)
            #pragma unroll
            for (int ns = 0; ns < 2; ++ns)
                acc[ms][ns] = __builtin_amdgcn_wmma_f32_16x16x32_bf16(
                    false, a[ms].v, false, b[ns].v, (short)0, acc[ms][ns],
                    false, false);

        stage_commit();   // next buffer ready; all waves done reading this one
        buf ^= 1;
    }

    #pragma unroll
    for (int ms = 0; ms < 4; ++ms)
        #pragma unroll
        for (int ns = 0; ns < 2; ++ns) {
            const int col = waveN + ns * 16 + ml;
            #pragma unroll
            for (int v = 0; v < 8; ++v) {
                const int row = waveM + ms * 16 + v + 8 * half;
                const float val = acc[ms][ns][v];
                if (c_is_f32)
                    ((float*)C)[(size_t)row * N + col] = val;
                else
                    ((__bf16*)C)[(size_t)row * N + col] = (__bf16)val;
            }
        }
}

// ---------------------------------------------------------------------------
// Kernel 3: RoPE on Q/K + split into attention layouts.
//   qkv bf16 [B*S, 6144]  (q | k | v blocks of 2048, head-major inside)
//   Qr [BH, S, 128], Kt [BH, 128, S] (transposed!), Vr [BH, S, 128]
// K/V scattered to cache index positions[s] (positions = arange -> identity).
// ---------------------------------------------------------------------------
__global__ void rope_split(const __bf16* __restrict__ qkv,
                           const int* __restrict__ positions,
                           __bf16* __restrict__ Qr, __bf16* __restrict__ Kt,
                           __bf16* __restrict__ Vr) {
    const int idx = blockIdx.x * blockDim.x + threadIdx.x;   // < B*S*H*64
    const int total = B_ * S_ * H_ * (HD_ / 2);
    if (idx >= total) return;
    const int pair = idx & 63;
    int t = idx >> 6;
    const int h = t & (H_ - 1);
    t >>= 4;
    const int s = t & (S_ - 1);
    const int b = t >> 11;

    const size_t row = ((size_t)b * S_ + s) * N3_;
    const int d0 = h * HD_ + pair * 2;

    const float q0 = (float)qkv[row + d0];
    const float q1 = (float)qkv[row + d0 + 1];
    const float k0 = (float)qkv[row + D_ + d0];
    const float k1 = (float)qkv[row + D_ + d0 + 1];
    const float v0 = (float)qkv[row + 2 * D_ + d0];
    const float v1 = (float)qkv[row + 2 * D_ + d0 + 1];

    const int   pos = positions[s];
    const float inv = __powf(10000.0f, -(float)(2 * pair) / (float)HD_);
    const float ang = (float)pos * inv;
    float sn, cs;
    __sincosf(ang, &sn, &cs);

    const float qe = q0 * cs - q1 * sn;
    const float qo = q1 * cs + q0 * sn;
    const float ke = k0 * cs - k1 * sn;
    const float ko = k1 * cs + k0 * sn;

    const size_t bh = (size_t)b * H_ + h;
    __bf16* qp = Qr + (bh * S_ + s) * HD_ + pair * 2;
    qp[0] = (__bf16)qe;
    qp[1] = (__bf16)qo;

    Kt[(bh * HD_ + pair * 2) * S_ + pos]     = (__bf16)ke;
    Kt[(bh * HD_ + pair * 2 + 1) * S_ + pos] = (__bf16)ko;

    __bf16* vp = Vr + (bh * S_ + (size_t)pos) * HD_ + pair * 2;
    vp[0] = (__bf16)v0;
    vp[1] = (__bf16)v1;
}

// ---------------------------------------------------------------------------
// Kernel 4: flash attention. Block = 128 threads = 4 waves; wave owns 16 q
// rows (block covers 64). K (transposed) and V tiles are staged once per
// 32-key block into LDS via async copies and shared by all 4 waves. Online
// softmax with wave32 __shfl_xor row reductions; P goes through LDS to turn
// the WMMA C-layout into an A-layout fragment. Output bf16 in [B, S, H*hd].
// ---------------------------------------------------------------------------
__global__ __launch_bounds__(128)
void flash_attn_wmma(const __bf16* __restrict__ Qr, const __bf16* __restrict__ Kt,
                     const __bf16* __restrict__ Vr, __bf16* __restrict__ Y) {
    __shared__ __align__(16) __bf16 sK[128 * 32];      // [dim][key]  8KB
    __shared__ __align__(16) __bf16 sV[32 * 128];      // [key][dim]  8KB
    __shared__ __align__(16) __bf16 plds[4][16 * 32];  // per-wave P  4KB

    const int lane = threadIdx.x & 31;
    const int wave = threadIdx.x >> 5;
    const int half = lane >> 4;
    const int ml   = lane & 15;

    const int qblk = blockIdx.x;                  // 0..31 (64 rows each)
    const int bh   = blockIdx.y;                  // 0..63
    const int q0   = qblk * 64 + wave * 16;       // wave's first q row

    const __bf16* Qp  = Qr + (size_t)bh * S_ * HD_;
    const __bf16* Ktp = Kt + (size_t)bh * HD_ * S_;
    const __bf16* Vp  = Vr + (size_t)bh * S_ * HD_;

    // Q fragments: 4 chunks of 32 along hd (kept in registers for all steps)
    FragB16 qf[4];
    #pragma unroll
    for (int c = 0; c < 4; ++c) {
        const __bf16* ap = Qp + (size_t)(q0 + ml) * HD_ + c * 32 + half * 8;
        qf[c].u[0] = *(const uint4*)(ap);
        qf[c].u[1] = *(const uint4*)(ap + 16);
    }

    v8f o[8];
    #pragma unroll
    for (int i = 0; i < 8; ++i) o[i] = zero8();
    float mrow[8], lrow[8];
    #pragma unroll
    for (int v = 0; v < 8; ++v) { mrow[v] = -1e30f; lrow[v] = 0.0f; }

    const float scale = 0.08838834764831845f;     // 1/sqrt(128)
    const int   jend  = qblk * 2 + 2;             // 32-key blocks (causal bound)

    for (int j = 0; j < jend; ++j) {
        const int key0 = j * 32;

        // ---- stage K tile (128x32) + V tile (32x128), shared by all waves ----
        #pragma unroll
        for (int i = 0; i < 4; ++i) {
            const int chunk = threadIdx.x + i * 128;   // 0..511
            const int krow = chunk >> 2, kc = chunk & 3;
            cp16_g2l(Ktp + (size_t)krow * S_ + key0 + kc * 8,
                     &sK[krow * 32 + kc * 8]);
            const int vrow = chunk >> 4, vc = chunk & 15;
            cp16_g2l(Vp + (size_t)(key0 + vrow) * HD_ + vc * 8,
                     &sV[vrow * 128 + vc * 8]);
        }
        stage_commit();

        // ---- S = Q K^T (16 x 32 tile as two 16x16 accumulators) ----
        v8f sacc[2];
        sacc[0] = zero8();
        sacc[1] = zero8();
        #pragma unroll
        for (int c = 0; c < 4; ++c) {
            #pragma unroll
            for (int kn = 0; kn < 2; ++kn) {
                FragB16 kf;
                const __bf16* bp = &sK[(c * 32) * 32 + kn * 16 + ml];
                #pragma unroll
                for (int s = 0; s < 16; ++s)
                    kf.v[s] = bp[((s & 7) + half * 8 + (s >> 3) * 16) * 32];
                sacc[kn] = __builtin_amdgcn_wmma_f32_16x16x32_bf16(
                    false, qf[c].v, false, kf.v, (short)0, sacc[kn], false, false);
            }
        }

        // ---- scale + causal mask + online softmax ----
        float newm[8];
        #pragma unroll
        for (int v = 0; v < 8; ++v) {
            const int row = q0 + v + 8 * half;
            float e0 = sacc[0][v] * scale;
            float e1 = sacc[1][v] * scale;
            if (key0 + ml > row)      e0 = -1e30f;
            if (key0 + 16 + ml > row) e1 = -1e30f;
            sacc[0][v] = e0;
            sacc[1][v] = e1;
            float mx = fmaxf(e0, e1);
            #pragma unroll
            for (int off = 1; off < 16; off <<= 1)
                mx = fmaxf(mx, __shfl_xor(mx, off, 32));
            newm[v] = fmaxf(mrow[v], mx);
        }
        #pragma unroll
        for (int v = 0; v < 8; ++v) {
            const float p0 = __expf(sacc[0][v] - newm[v]);
            const float p1 = __expf(sacc[1][v] - newm[v]);
            sacc[0][v] = p0;
            sacc[1][v] = p1;
            float rs = p0 + p1;
            #pragma unroll
            for (int off = 1; off < 16; off <<= 1)
                rs += __shfl_xor(rs, off, 32);
            const float alpha = __expf(mrow[v] - newm[v]);
            lrow[v] = lrow[v] * alpha + rs;
            mrow[v] = newm[v];
            #pragma unroll
            for (int t = 0; t < 8; ++t) o[t][v] = o[t][v] * alpha;
        }

        // ---- P: C-layout -> LDS (row-major 16x32 bf16) -> A-layout frag ----
        __bf16* pl = plds[wave];
        #pragma unroll
        for (int v = 0; v < 8; ++v) {
            const int row = v + 8 * half;
            pl[row * 32 + ml]      = (__bf16)sacc[0][v];
            pl[row * 32 + 16 + ml] = (__bf16)sacc[1][v];
        }
        __syncthreads();
        FragB16 pf;
        {
            const __bf16* ap = pl + ml * 32 + half * 8;
            pf.u[0] = *(const uint4*)(ap);
            pf.u[1] = *(const uint4*)(ap + 16);
        }
        __syncthreads();

        // ---- O += P @ V (V fragments from LDS) ----
        #pragma unroll
        for (int t = 0; t < 8; ++t) {
            FragB16 vf;
            const __bf16* bp = &sV[t * 16 + ml];
            #pragma unroll
            for (int s = 0; s < 16; ++s)
                vf.v[s] = bp[((s & 7) + half * 8 + (s >> 3) * 16) * 128];
            o[t] = __builtin_amdgcn_wmma_f32_16x16x32_bf16(
                false, pf.v, false, vf.v, (short)0, o[t], false, false);
        }
        __syncthreads();   // sK/sV consumed; safe to restage next iteration
    }

    // ---- normalize and write y in [B, S, H*hd] (bf16) ----
    const int b = bh >> 4;
    const int h = bh & 15;
    #pragma unroll
    for (int v = 0; v < 8; ++v) {
        const int s = q0 + v + 8 * half;
        const float inv_l = 1.0f / lrow[v];
        __bf16* yp = Y + ((size_t)(b * S_ + s)) * D_ + h * HD_ + ml;
        #pragma unroll
        for (int t = 0; t < 8; ++t)
            yp[t * 16] = (__bf16)(o[t][v] * inv_l);
    }
}

// ---------------------------------------------------------------------------
// Host-side orchestration
// ---------------------------------------------------------------------------
extern "C" void kernel_launch(void* const* d_in, const int* in_sizes, int n_in,
                              void* d_out, int out_size, void* d_ws, size_t ws_size,
                              hipStream_t stream) {
    const float* x         = (const float*)d_in[0];
    const int*   positions = (const int*)d_in[1];
    // d_in[2] = mask (bool) -- causality computed analytically, unused.
    const float* Wqkv      = (const float*)d_in[3];
    const float* Wout      = (const float*)d_in[4];
    float*       out       = (float*)d_out;

    char* ws = (char*)d_ws;
    size_t off = 0;
    __bf16* Xb    = (__bf16*)(ws + off); off += (size_t)M_ * D_ * 2;       // 32MB
    __bf16* Wqkvb = (__bf16*)(ws + off); off += (size_t)D_ * N3_ * 2;      // 24MB
    __bf16* Woutb = (__bf16*)(ws + off); off += (size_t)D_ * D_ * 2;       //  8MB
    __bf16* qkvb  = (__bf16*)(ws + off); off += (size_t)M_ * N3_ * 2;      // 96MB
    __bf16* Qr    = (__bf16*)(ws + off); off += (size_t)BH_ * S_ * HD_ * 2;// 32MB
    __bf16* Kt    = (__bf16*)(ws + off); off += (size_t)BH_ * HD_ * S_ * 2;// 32MB
    __bf16* Vr    = (__bf16*)(ws + off); off += (size_t)BH_ * S_ * HD_ * 2;// 32MB
    __bf16* Yb    = (__bf16*)(ws + off); off += (size_t)M_ * D_ * 2;       // 32MB

    // 1) fp32 -> bf16
    cvt_f32_to_bf16<<<4096, 256, 0, stream>>>(x, Xb, (size_t)M_ * D_);
    cvt_f32_to_bf16<<<4096, 256, 0, stream>>>(Wqkv, Wqkvb, (size_t)D_ * N3_);
    cvt_f32_to_bf16<<<4096, 256, 0, stream>>>(Wout, Woutb, (size_t)D_ * D_);

    // 2) qkv = X @ Wqkv   [8192,2048] x [2048,6144] -> bf16
    gemm_bf16_wmma<<<dim3(N3_ / 128, M_ / 128), 256, 0, stream>>>(
        Xb, Wqkvb, qkvb, M_, N3_, D_, 0);

    // 3) RoPE + split (K transposed)
    {
        const int total = B_ * S_ * H_ * (HD_ / 2);
        rope_split<<<(total + 255) / 256, 256, 0, stream>>>(
            qkvb, positions, Qr, Kt, Vr);
    }

    // 4) flash attention -> Yb bf16 [B, S, D]
    flash_attn_wmma<<<dim3(S_ / 64, BH_), 128, 0, stream>>>(Qr, Kt, Vr, Yb);

    // 5) out = Y @ Wout   [8192,2048] x [2048,2048] -> f32
    gemm_bf16_wmma<<<dim3(D_ / 128, M_ / 128), 256, 0, stream>>>(
        Yb, Woutb, out, M_, D_, D_, 1);

    (void)in_sizes; (void)n_in; (void)out_size; (void)ws_size;
}
